// LinearCrossAttention_15590731284899
// MI455X (gfx1250) — matrix operand order repaired and checked by reading, not compiled, not run on hardware
//
#include <hip/hip_runtime.h>
#include <hip/hip_bf16.h>

#define B_  4
#define NQ_ 4096
#define NC_ 4096
#define D_  1024
#define H_  16
#define HD_ 64

typedef __attribute__((ext_vector_type(16))) _Float16 v16h;
typedef __attribute__((ext_vector_type(8)))  _Float16 v8h;
typedef __attribute__((ext_vector_type(4)))  _Float16 v4h;
typedef __attribute__((ext_vector_type(8)))  float    v8f;

union Frag16 { v16h v; v8h h8[2]; _Float16 e[16]; };
union Acc8   { v8f  v; float f[8]; };
union H8     { v8h  v; _Float16 e[8]; };

// ---------------------------------------------------------------------------
// Kernel 1: row LayerNorm (fp32 math) -> f16.  256 thr, 4 elems/thr (D=1024)
// ---------------------------------------------------------------------------
__global__ void ln_to_f16(const float* __restrict__ x, const float* __restrict__ g,
                          const float* __restrict__ bta, _Float16* __restrict__ y) {
  const int row = blockIdx.x;
  const int t   = threadIdx.x;
  const float4 xv = *(const float4*)(x + (size_t)row * D_ + t * 4);
  float s  = xv.x + xv.y + xv.z + xv.w;
  float s2 = xv.x * xv.x + xv.y * xv.y + xv.z * xv.z + xv.w * xv.w;
  __shared__ float rs[8], rs2[8];
  #pragma unroll
  for (int o = 16; o > 0; o >>= 1) { s += __shfl_down(s, o, 32); s2 += __shfl_down(s2, o, 32); }
  const int wave = t >> 5, lane = t & 31;
  if (lane == 0) { rs[wave] = s; rs2[wave] = s2; }
  __syncthreads();
  __shared__ float mu_s, inv_s;
  if (t == 0) {
    float t1 = 0.f, t2 = 0.f;
    #pragma unroll
    for (int i = 0; i < 8; ++i) { t1 += rs[i]; t2 += rs2[i]; }
    const float mu  = t1 * (1.f / D_);
    const float var = t2 * (1.f / D_) - mu * mu;
    mu_s = mu; inv_s = rsqrtf(var + 1e-5f);
  }
  __syncthreads();
  const float mu = mu_s, inv = inv_s;
  const float4 gv = *(const float4*)(g + t * 4);
  const float4 bv = *(const float4*)(bta + t * 4);
  union { v4h v; _Float16 e[4]; } o;
  o.e[0] = (_Float16)((xv.x - mu) * inv * gv.x + bv.x);
  o.e[1] = (_Float16)((xv.y - mu) * inv * gv.y + bv.y);
  o.e[2] = (_Float16)((xv.z - mu) * inv * gv.z + bv.z);
  o.e[3] = (_Float16)((xv.w - mu) * inv * gv.w + bv.w);
  *(v4h*)(y + (size_t)row * D_ + t * 4) = o.v;
}

// ---------------------------------------------------------------------------
// Kernel 2: fp32 -> f16 (weights), 8 elems/thread
// ---------------------------------------------------------------------------
__global__ void f32_to_f16(const float* __restrict__ x, _Float16* __restrict__ y) {
  const int i = blockIdx.x * 256 + threadIdx.x;
  const float4 a = *(const float4*)(x + (size_t)i * 8);
  const float4 b = *(const float4*)(x + (size_t)i * 8 + 4);
  H8 o;
  o.e[0] = (_Float16)a.x; o.e[1] = (_Float16)a.y; o.e[2] = (_Float16)a.z; o.e[3] = (_Float16)a.w;
  o.e[4] = (_Float16)b.x; o.e[5] = (_Float16)b.y; o.e[6] = (_Float16)b.z; o.e[7] = (_Float16)b.w;
  *(v8h*)(y + (size_t)i * 8) = o.v;
}

// ---------------------------------------------------------------------------
// Kernel 3/6: WMMA GEMM  Y = X @ W^T (+bias).
// Whole 64x1024 W panel staged ONCE in LDS (132KB, 2 WG/WGP); A fragments
// stream straight from global; K-loop barrier-free; 2 M-tiles per wave
// (8 WMMA per 12 x 16B loads).
//   MODE 0: Q: elu(y)+1 -> f16 (b,h,nq,hd)   [epilogue staged in dead Bs]
//   MODE 1: K: elu(y)+1 -> f16 (b,h,hd,nc)   [direct packed v8h stores]
//   MODE 2: V: linear   -> f16 (b,h,hd,nc)   [direct packed v8h stores]
//   MODE 3: O: y + residual -> fp32 (b,n,D)  [epilogue staged in dead Bs]
// Block 256 thr = 8 waves; BM=256, BN=64.
// ---------------------------------------------------------------------------
template<int MODE>
__global__ void proj_gemm(const _Float16* __restrict__ A,    // (Mtot, D) f16
                          const _Float16* __restrict__ W,    // (D, D) f16 (n,k)
                          const float* __restrict__ bias,
                          _Float16* __restrict__ outH,
                          const float* __restrict__ resid,
                          float* __restrict__ outF) {
  constexpr int BM = 256, BN = 64, LDB = 1032;   // 1032 halves: 4-bank skew
  __shared__ _Float16 Bs[BN][LDB];               // (n, k) full panel, 132KB
  const int m0 = blockIdx.x * BM;
  const int n0 = blockIdx.y * BN;
  const int t  = threadIdx.x;
  const int wave = t >> 5, lane = t & 31, lm = lane & 15, lg = lane >> 4;

  // stage entire W panel once: thread t copies 256 contiguous halves
  {
    const int row = t >> 2, kc = (t & 3) * 256;
    const _Float16* src = W + (size_t)(n0 + row) * D_ + kc;
    #pragma unroll
    for (int j = 0; j < 32; ++j)
      *(v8h*)&Bs[row][kc + j * 8] = *(const v8h*)(src + j * 8);
  }
  __syncthreads();

  // barrier-free K loop: A direct from global, B from resident LDS panel.
  // Each wave owns 32 rows (two 16-row M-tiles) -> B frags amortized 2x.
  const _Float16* arow0 = A + (size_t)(m0 + wave * 32 + lm) * D_;
  const _Float16* arow1 = arow0 + (size_t)16 * D_;
  Acc8 acc[2][4] = {};
  for (int k0 = 0; k0 < D_; k0 += 32) {
    Frag16 af0, af1;   // A-layout: 16 contiguous at k0+16*lg
    af0.h8[0] = *(const v8h*)(arow0 + k0 + lg * 16);
    af0.h8[1] = *(const v8h*)(arow0 + k0 + lg * 16 + 8);
    af1.h8[0] = *(const v8h*)(arow1 + k0 + lg * 16);
    af1.h8[1] = *(const v8h*)(arow1 + k0 + lg * 16 + 8);
    #pragma unroll
    for (int nt = 0; nt < 4; ++nt) {
      Frag16 bf;
      bf.h8[0] = *(const v8h*)&Bs[nt * 16 + lm][k0 + lg * 16];
      bf.h8[1] = *(const v8h*)&Bs[nt * 16 + lm][k0 + lg * 16 + 8];
      acc[0][nt].v = __builtin_amdgcn_wmma_f32_16x16x32_f16(
          false, af0.v, false, bf.v, (short)0, acc[0][nt].v, false, false);
      acc[1][nt].v = __builtin_amdgcn_wmma_f32_16x16x32_f16(
          false, af1.v, false, bf.v, (short)0, acc[1][nt].v, false, false);
    }
  }

  const int b  = m0 >> 12;                 // batch (block never straddles)
  const int hh = n0 >> 6;                  // head  (BN == HD)

  if constexpr (MODE == 0) {
    // stage f16 C tile in dead Bs memory, then packed (b,h,nq,hd) stores
    __syncthreads();                       // all b-frag reads done
    _Float16* Cs = (_Float16*)&Bs[0][0];   // [256][72]
    #pragma unroll
    for (int mm = 0; mm < 2; ++mm) {
      #pragma unroll
      for (int nt = 0; nt < 4; ++nt) {
        const float bv = bias[n0 + nt * 16 + lm];
        #pragma unroll
        for (int r = 0; r < 8; ++r) {
          float v = acc[mm][nt].f[r] + bv;
          v = (v > 0.f) ? (v + 1.f) : __expf(v);        // elu(x)+1
          Cs[(wave * 32 + mm * 16 + r + 8 * lg) * 72 + nt * 16 + lm] = (_Float16)v;
        }
      }
    }
    __syncthreads();
    const int nq = (m0 & (NQ_ - 1)) + t;   // one full row per thread
    _Float16* dst = outH + ((size_t)(b * H_ + hh) * NQ_ + nq) * HD_;
    #pragma unroll
    for (int c2 = 0; c2 < 8; ++c2)
      *(v8h*)(dst + c2 * 8) = *(const v8h*)&Cs[t * 72 + c2 * 8];
  } else if constexpr (MODE <= 2) {
    // transposed (b,h,hd,nc): consecutive r -> consecutive nc -> one v8h
    #pragma unroll
    for (int mm = 0; mm < 2; ++mm) {
      const int nq0 = (m0 & (NC_ - 1)) + wave * 32 + mm * 16 + 8 * lg;
      #pragma unroll
      for (int nt = 0; nt < 4; ++nt) {
        const int gn = n0 + nt * 16 + lm;
        const float bv = bias[gn];
        H8 o;
        #pragma unroll
        for (int r = 0; r < 8; ++r) {
          float v = acc[mm][nt].f[r] + bv;
          if (MODE == 1) v = (v > 0.f) ? (v + 1.f) : __expf(v);
          o.e[r] = (_Float16)v;
        }
        *(v8h*)&outH[((size_t)(b * H_ + hh) * HD_ + (gn & (HD_ - 1))) * NC_ + nq0] = o.v;
      }
    }
  } else {
    // stage fp32 C tile in dead Bs memory, coalesced residual-add epilogue
    __syncthreads();
    float* Cs3 = (float*)&Bs[0][0];        // [256][68]
    #pragma unroll
    for (int mm = 0; mm < 2; ++mm) {
      #pragma unroll
      for (int nt = 0; nt < 4; ++nt) {
        const float bv = bias[n0 + nt * 16 + lm];
        #pragma unroll
        for (int r = 0; r < 8; ++r)
          Cs3[(wave * 32 + mm * 16 + r + 8 * lg) * 68 + nt * 16 + lm] =
              acc[mm][nt].f[r] + bv;
      }
    }
    __syncthreads();
    const float* rsrc = resid + (size_t)(m0 + t) * D_ + n0;   // one row/thread
    float*       dstF = outF  + (size_t)(m0 + t) * D_ + n0;
    #pragma unroll
    for (int c2 = 0; c2 < 16; ++c2) {
      const float4 rv = *(const float4*)(rsrc + c2 * 4);
      const float4 cv = *(const float4*)&Cs3[t * 68 + c2 * 4];
      float4 ov; ov.x = rv.x + cv.x; ov.y = rv.y + cv.y;
      ov.z = rv.z + cv.z; ov.w = rv.w + cv.w;
      *(float4*)(dstF + c2 * 4) = ov;
    }
  }
}

// ---------------------------------------------------------------------------
// Kernel 4: per (b,h)  KVt[e][d] = (K^T V)^T, K_sum[d].
// K,V stored transposed (hd, nc): fragments load straight from global.
// Barrier-free; 6 x v8h loads + 2 wmma per 32-chunk per wave.
// ---------------------------------------------------------------------------
__global__ void kv_kernel(const _Float16* __restrict__ Kt, const _Float16* __restrict__ Vt,
                          _Float16* __restrict__ KVt, float* __restrict__ Ksum) {
  const int bh = blockIdx.x;
  const _Float16* Kb = Kt + (size_t)bh * HD_ * NC_;   // (d, n)
  const _Float16* Vb = Vt + (size_t)bh * HD_ * NC_;   // (e, n)
  const int t = threadIdx.x;

  // ---- K_sum: contiguous row sums, 4 threads per row + shuffle reduce ----
  {
    const int d = t >> 2, part = t & 3;
    const _Float16* row = Kb + (size_t)d * NC_;
    float s = 0.f;
    for (int n = part * 8; n < NC_; n += 32) {
      H8 q; q.v = *(const v8h*)(row + n);
      #pragma unroll
      for (int j = 0; j < 8; ++j) s += (float)q.e[j];
    }
    s += __shfl_down(s, 2, 32);
    s += __shfl_down(s, 1, 32);
    if (part == 0) Ksum[bh * 64 + d] = s;
  }

  // ---- KV = K^T @ V ----
  const int wave = t >> 5, lane = t & 31, lm = lane & 15, lg = lane >> 4;
  const int mt = wave >> 1;                 // 8 waves cover 4x4 tiles, 2 each
  const int ntb = 2 * (wave & 1);
  const _Float16* arow  = Kb + (size_t)(mt * 16 + lm) * NC_;
  const _Float16* brow0 = Vb + (size_t)(ntb * 16 + lm) * NC_;
  const _Float16* brow1 = brow0 + (size_t)16 * NC_;
  Acc8 acc[2] = {};
  for (int c = 0; c < NC_; c += 32) {
    Frag16 af;   // A-layout: 16 contiguous at c+16*lg
    af.h8[0] = *(const v8h*)(arow + c + lg * 16);
    af.h8[1] = *(const v8h*)(arow + c + lg * 16 + 8);
    Frag16 b0;   // B-layout: 16 contiguous at c+16*lg
    b0.h8[0] = *(const v8h*)(brow0 + c + lg * 16);
    b0.h8[1] = *(const v8h*)(brow0 + c + lg * 16 + 8);
    acc[0].v = __builtin_amdgcn_wmma_f32_16x16x32_f16(
        false, af.v, false, b0.v, (short)0, acc[0].v, false, false);
    Frag16 b1;
    b1.h8[0] = *(const v8h*)(brow1 + c + lg * 16);
    b1.h8[1] = *(const v8h*)(brow1 + c + lg * 16 + 8);
    acc[1].v = __builtin_amdgcn_wmma_f32_16x16x32_f16(
        false, af.v, false, b1.v, (short)0, acc[1].v, false, false);
  }
  // store (e,d): consecutive r -> consecutive d -> one v8h store
  #pragma unroll
  for (int s = 0; s < 2; ++s) {
    const int nt = ntb + s;
    H8 o;
    #pragma unroll
    for (int r = 0; r < 8; ++r) o.e[r] = (_Float16)acc[s].f[r];
    *(v8h*)&KVt[((size_t)bh * 64 + nt * 16 + lm) * 64 + mt * 16 + 8 * lg] = o.v;
  }
}

// ---------------------------------------------------------------------------
// Kernel 5: per (b,h): out = (Q @ KV) / max(Q . K_sum, 1e-6), f16 -> (B,N,D)
// ---------------------------------------------------------------------------
__global__ void out_kernel(const _Float16* __restrict__ Qf, const _Float16* __restrict__ KVt,
                           const float* __restrict__ Ksum, _Float16* __restrict__ attn) {
  const int bh = blockIdx.y;
  const int m0 = blockIdx.x * 128;
  const int t  = threadIdx.x;
  const _Float16* Qb = Qf + (size_t)bh * NQ_ * HD_;
  __shared__ _Float16 KVs[64][72];     // (e, d)
  __shared__ float kss[64];
  __shared__ float Zs[128];
  {
    const int e = t >> 2, dbase = (t & 3) * 16;
    const _Float16* src = KVt + (size_t)bh * 4096 + (size_t)e * 64 + dbase;
    *(v8h*)&KVs[e][dbase]     = *(const v8h*)(src);
    *(v8h*)&KVs[e][dbase + 8] = *(const v8h*)(src + 8);
  }
  if (t < 64) kss[t] = Ksum[bh * 64 + t];
  __syncthreads();

  if (t < 128) {
    const _Float16* qrow = Qb + (size_t)(m0 + t) * HD_;
    float z = 0.f;
    #pragma unroll
    for (int c = 0; c < 8; ++c) {
      H8 q; q.v = *(const v8h*)(qrow + c * 8);
      #pragma unroll
      for (int j = 0; j < 8; ++j) z += (float)q.e[j] * kss[c * 8 + j];
    }
    Zs[t] = fmaxf(z, 1e-6f);
  }
  __syncthreads();

  const int wave = t >> 5, lane = t & 31, lm = lane & 15, lg = lane >> 4;
  const int mrow = m0 + wave * 16;
  const _Float16* qrow = Qb + (size_t)(mrow + lm) * HD_;
  Acc8 acc[4] = {};
  #pragma unroll
  for (int k0 = 0; k0 < 64; k0 += 32) {
    Frag16 af;   // A fragment straight from global: two contiguous 8-chunks
    af.h8[0] = *(const v8h*)(qrow + k0 + lg * 16);
    af.h8[1] = *(const v8h*)(qrow + k0 + lg * 16 + 8);
    #pragma unroll
    for (int nt = 0; nt < 4; ++nt) {
      Frag16 bf;
      bf.h8[0] = *(const v8h*)&KVs[nt * 16 + lm][k0 + lg * 16];
      bf.h8[1] = *(const v8h*)&KVs[nt * 16 + lm][k0 + lg * 16 + 8];
      acc[nt].v = __builtin_amdgcn_wmma_f32_16x16x32_f16(
          false, af.v, false, bf.v, (short)0, acc[nt].v, false, false);
    }
  }
  // stage f16 tile then packed coalesced stores into (b, nq, D)
  __shared__ _Float16 Cs[128][72];
  #pragma unroll
  for (int nt = 0; nt < 4; ++nt) {
    #pragma unroll
    for (int r = 0; r < 8; ++r) {
      const int M = r + 8 * lg;
      Cs[wave * 16 + M][nt * 16 + lm] = (_Float16)(acc[nt].f[r] / Zs[wave * 16 + M]);
    }
  }
  __syncthreads();
  const int b = bh >> 4, hh = bh & 15;
  const int row = t >> 1, col = (t & 1) * 32;
  _Float16* dst = attn + ((size_t)b * NQ_ + m0 + row) * D_ + hh * HD_ + col;
  #pragma unroll
  for (int c2 = 0; c2 < 4; ++c2)
    *(v8h*)(dst + c2 * 8) = *(const v8h*)&Cs[row][col + c2 * 8];
}

// ---------------------------------------------------------------------------
// Host launcher
// ---------------------------------------------------------------------------
extern "C" void kernel_launch(void* const* d_in, const int* in_sizes, int n_in,
                              void* d_out, int out_size, void* d_ws, size_t ws_size,
                              hipStream_t stream) {
  (void)in_sizes; (void)n_in; (void)out_size; (void)ws_size;
  const float* query   = (const float*)d_in[0];
  const float* context = (const float*)d_in[1];
  const float* q_w = (const float*)d_in[2];  const float* q_b = (const float*)d_in[3];
  const float* k_w = (const float*)d_in[4];  const float* k_b = (const float*)d_in[5];
  const float* v_w = (const float*)d_in[6];  const float* v_b = (const float*)d_in[7];
  const float* o_w = (const float*)d_in[8];  const float* o_b = (const float*)d_in[9];
  const float* lnq_g  = (const float*)d_in[10]; const float* lnq_b  = (const float*)d_in[11];
  const float* lnkv_g = (const float*)d_in[12]; const float* lnkv_b = (const float*)d_in[13];
  float* out = (float*)d_out;

  char* ws = (char*)d_ws;
  const size_t SQ = (size_t)B_ * NQ_ * D_;          // 16M elements
  _Float16* qn   = (_Float16*)(ws + 0 * SQ);        // 32MB (reused as attn later)
  _Float16* cn   = (_Float16*)(ws + 2 * SQ);        // 32MB
  _Float16* Qf   = (_Float16*)(ws + 4 * SQ);        // 32MB (b,h,nq,hd)
  _Float16* Kf   = (_Float16*)(ws + 6 * SQ);        // 32MB (b,h,hd,nc)
  _Float16* Vf   = (_Float16*)(ws + 8 * SQ);        // 32MB (b,h,hd,nc)
  _Float16* w16  = (_Float16*)(ws + 10 * SQ);       // 4 x 2MB f16 weights
  const size_t WN = (size_t)D_ * D_;
  _Float16* wq16 = w16;
  _Float16* wk16 = w16 + WN;
  _Float16* wv16 = w16 + 2 * WN;
  _Float16* wo16 = w16 + 3 * WN;
  _Float16* kvt  = (_Float16*)(ws + 10 * SQ + 8 * WN);          // 512KB
  float*    ksum = (float*)(ws + 10 * SQ + 8 * WN +
                            (size_t)B_ * H_ * HD_ * HD_ * 2);   // 16KB
  _Float16* attn = qn;   // safe alias: qn consumed before attn written

  // 1) LayerNorms -> f16
  ln_to_f16<<<dim3(B_ * NQ_), 256, 0, stream>>>(query,   lnq_g,  lnq_b,  qn);
  ln_to_f16<<<dim3(B_ * NC_), 256, 0, stream>>>(context, lnkv_g, lnkv_b, cn);

  // 2) Weights -> f16 (8 elems/thread)
  const int wg = (int)(WN / (256 * 8));
  f32_to_f16<<<wg, 256, 0, stream>>>(q_w, wq16);
  f32_to_f16<<<wg, 256, 0, stream>>>(k_w, wk16);
  f32_to_f16<<<wg, 256, 0, stream>>>(v_w, wv16);
  f32_to_f16<<<wg, 256, 0, stream>>>(o_w, wo16);

  // 3) Projections: Q row-major; K,V head-transposed; fused elu+1 for Q,K
  dim3 pg(B_ * NQ_ / 256, D_ / 64);
  proj_gemm<0><<<pg, 256, 0, stream>>>(qn, wq16, q_b, Qf, nullptr, nullptr);
  proj_gemm<1><<<pg, 256, 0, stream>>>(cn, wk16, k_b, Kf, nullptr, nullptr);
  proj_gemm<2><<<pg, 256, 0, stream>>>(cn, wv16, v_b, Vf, nullptr, nullptr);

  // 4) Per-head KV^T and K_sum (barrier-free WMMA reduction)
  kv_kernel<<<dim3(B_ * H_), 256, 0, stream>>>(Kf, Vf, kvt, ksum);

  // 5) Per-head (Q @ KV) / Z  -> attn (B,N,D) f16
  out_kernel<<<dim3(NQ_ / 128, B_ * H_), 256, 0, stream>>>(Qf, kvt, ksum, attn);

  // 6) Output projection + bias + residual -> fp32 out
  proj_gemm<3><<<pg, 256, 0, stream>>>(attn, wo16, o_b, nullptr, query, out);
}
